// DualScaleVQ_24902220382644
// MI455X (gfx1250) — matrix operand order repaired and checked by reading, not compiled, not run on hardware
//
#include <hip/hip_runtime.h>

typedef __attribute__((ext_vector_type(16))) __bf16 v16bf;
typedef __attribute__((ext_vector_type(8)))  float  v8f;

#define N_ROWS 32768
#define LD 256
#define DDIM 512
#define KSYN 1024
#define KSEM 4096
#define CT 32          // codes per LDS tile
#define LSTRIDE 520    // ushorts per code row in LDS (512 + 8 pad -> bank-conflict free)

__device__ __forceinline__ unsigned int pack2bf(float a, float b) {
  unsigned int ua = __float_as_uint(a);
  unsigned int ub = __float_as_uint(b);
  ua = (ua + 0x7FFFu + ((ua >> 16) & 1u)) >> 16;   // round-to-nearest-even bf16
  ub = (ub + 0x7FFFu + ((ub >> 16) & 1u)) >> 16;
  return (ua & 0xFFFFu) | (ub << 16);
}
__device__ __forceinline__ unsigned short bf16bits(float a) {
  unsigned int ua = __float_as_uint(a);
  ua = (ua + 0x7FFFu + ((ua >> 16) & 1u)) >> 16;
  return (unsigned short)ua;
}

union Frag {
  v16bf v;
  unsigned int u[8];
  uint4 q[2];
};

// ---------------------------------------------------------------- utilities
__global__ void zero_kernel(float* __restrict__ p, long long n) {
  long long i = (long long)blockIdx.x * blockDim.x + threadIdx.x;
  long long stride = (long long)gridDim.x * blockDim.x;
  for (; i < n; i += stride) p[i] = 0.0f;
}

// convert one codebook f32 -> bf16 (ws) and compute per-code squared norms
__global__ void cb_convert_kernel(const float* __restrict__ cb,
                                  unsigned short* __restrict__ cbbf,
                                  float* __restrict__ cnorm, int K) {
  int wave = threadIdx.x >> 5, lane = threadIdx.x & 31;
  int code = blockIdx.x * 8 + wave;
  if (code >= K) return;
  const float* src = cb + (size_t)code * DDIM;
  unsigned short* dst = cbbf + (size_t)code * DDIM;
  float s = 0.f;
  for (int t = lane; t < DDIM; t += 32) {
    float v = src[t];
    s += v * v;
    dst[t] = bf16bits(v);
  }
#pragma unroll
  for (int m = 16; m >= 1; m >>= 1) s += __shfl_xor(s, m, 32);
  if (lane == 0) cnorm[code] = s;
}

// ------------------------------------------------ WMMA distance + argmin
// block = 256 threads (8 waves), each wave owns 16 rows of z (A resident in regs)
__global__ __launch_bounds__(256) void vq_argmin_kernel(
    const float* __restrict__ zr, const float* __restrict__ zi,
    const unsigned short* __restrict__ cbbf,
    const float* __restrict__ cnorm, int K, int* __restrict__ idx_out) {
  __shared__ unsigned short sm[CT * LSTRIDE];
  const int lane = threadIdx.x & 31;
  const int wave = threadIdx.x >> 5;
  const int hi   = lane >> 4;        // which 16-lane half
  const int m16  = lane & 15;
  const int s    = hi * 8;           // A-layout K-offset for this half
  const int row  = blockIdx.x * 128 + wave * 16 + m16;

  // --- load 16 rows x 512 dims as 16 bf16 A-fragments (ISA 16-bit A layout) ---
  Frag a[16];
  const float* rpr = zr + (size_t)row * LD;
  const float* rpi = zi + (size_t)row * LD;
#pragma unroll
  for (int c0 = 0; c0 < 16; ++c0) {
    const float* rp = (c0 < 8) ? (rpr + c0 * 32) : (rpi + (c0 - 8) * 32);
#pragma unroll
    for (int v = 0; v < 8; ++v) {
      int kk = (v < 4 ? 2 * v : 16 + 2 * (v - 4)) + s;
      a[c0].u[v] = pack2bf(rp[kk], rp[kk + 1]);
    }
  }

  float minv[8];
  int mini[8];
#pragma unroll
  for (int r = 0; r < 8; ++r) { minv[r] = 3.4e38f; mini[r] = 0; }

  unsigned int* sm32 = (unsigned int*)sm;
  for (int ct = 0; ct < K; ct += CT) {
    __syncthreads();
    // stage CT codes (bf16) into padded LDS, coalesced dword copies
    const unsigned int* src = (const unsigned int*)cbbf + (size_t)ct * (DDIM / 2);
    for (int j = threadIdx.x; j < CT * (DDIM / 2); j += 256) {
      int code = j >> 8;          // DDIM/2 == 256 dwords per code
      int dp = j & 255;
      sm32[code * (LSTRIDE / 2) + dp] = src[code * (DDIM / 2) + dp];
    }
    __syncthreads();
#pragma unroll
    for (int sub = 0; sub < CT / 16; ++sub) {
      int codeLocal = sub * 16 + m16;
      // B layout: lane = N (code), K = hi*8 + [0..7] and 16 + hi*8 + [0..7]
      const unsigned short* brow = sm + codeLocal * LSTRIDE + hi * 8;
      v8f acc = {0.f, 0.f, 0.f, 0.f, 0.f, 0.f, 0.f, 0.f};
#pragma unroll
      for (int c0 = 0; c0 < 16; ++c0) {
        Frag b;
        b.q[0] = *(const uint4*)(brow + c0 * 32);
        b.q[1] = *(const uint4*)(brow + c0 * 32 + 16);
        acc = __builtin_amdgcn_wmma_f32_16x16x32_bf16(false, a[c0].v, false, b.v,
                                                      (short)0, acc, false, false);
      }
      int code = ct + codeLocal;
      float cn = cnorm[code];
#pragma unroll
      for (int r = 0; r < 8; ++r) {
        float sc = cn - 2.0f * acc[r];   // ||z||^2 constant per row -> dropped
        if (sc < minv[r]) { minv[r] = sc; mini[r] = code; }
      }
    }
  }
  // reduce min across the 16 columns of each half-wave (C layout: N = lane&15)
#pragma unroll
  for (int r = 0; r < 8; ++r) {
#pragma unroll
    for (int m = 1; m <= 8; m <<= 1) {
      float ov = __shfl_xor(minv[r], m, 32);
      int   oi = __shfl_xor(mini[r], m, 32);
      if (ov < minv[r] || (ov == minv[r] && oi < mini[r])) { minv[r] = ov; mini[r] = oi; }
    }
  }
  if (m16 == 0) {
    int rbase = blockIdx.x * 128 + wave * 16 + hi * 8;  // C layout: VGPR r -> M = r + 8*hi
#pragma unroll
    for (int r = 0; r < 8; ++r) idx_out[rbase + r] = mini[r];
  }
}

// ---------------- per-row: zq gather, loss accum, embed/count/pair scatters
__global__ void row_update_kernel(const float* __restrict__ zr, const float* __restrict__ zi,
                                  const float* __restrict__ cb,
                                  const int* __restrict__ idx, const int* __restrict__ prev,
                                  int K, float* __restrict__ zq, float* __restrict__ embed,
                                  float* __restrict__ counts, float* __restrict__ pair,
                                  float* __restrict__ losssum) {
  __shared__ float red[4];
  int i = blockIdx.x;
  int c = idx[i];
  float ls = 0.f;
  const float* cbr = cb + (size_t)c * DDIM;
  for (int t = threadIdx.x; t < DDIM; t += 128) {
    float zv = (t < LD) ? zr[(size_t)i * LD + t] : zi[(size_t)i * LD + t - LD];
    float q = cbr[t];
    zq[(size_t)i * DDIM + t] = q;   // straight-through: zq_st == z_q numerically
    float d = q - zv;
    ls += d * d;
    atomicAdd(&embed[(size_t)c * DDIM + t], zv);
  }
#pragma unroll
  for (int m = 16; m >= 1; m >>= 1) ls += __shfl_xor(ls, m, 32);
  int lane = threadIdx.x & 31, wv = threadIdx.x >> 5;
  if (lane == 0) red[wv] = ls;
  __syncthreads();
  if (threadIdx.x == 0) {
    atomicAdd(losssum, red[0] + red[1] + red[2] + red[3]);
    atomicAdd(&counts[c], 1.0f);
    atomicAdd(&pair[(size_t)prev[i] * K + c], 1.0f);
  }
}

__global__ void cl_kernel(const float* __restrict__ cl, const float* __restrict__ counts,
                          float* __restrict__ clnew, float* __restrict__ nsum, int K) {
  int k = blockIdx.x * blockDim.x + threadIdx.x;
  if (k < K) {
    float v = cl[k] * 0.99f + 0.01f * counts[k];
    clnew[k] = v;
    atomicAdd(nsum, v);
  }
}

__global__ void cb_final_kernel(const float* __restrict__ avg, const float* __restrict__ clnew,
                                const float* __restrict__ nsum, float* __restrict__ cbout, int K) {
  long long gid = (long long)blockIdx.x * blockDim.x + threadIdx.x;
  if (gid < (long long)K * DDIM) {
    int k = (int)(gid >> 9);
    float n = nsum[0];
    float cs = (clnew[k] + 1e-6f) / (n + (float)K * 1e-6f) * n;
    float av = avg[gid] * 0.99f + 0.01f * cbout[gid];  // cbout currently holds embed_sum
    cbout[gid] = av / cs;
  }
}

__global__ void adj_final_kernel(const float* __restrict__ adj, float* __restrict__ out,
                                 long long n) {
  long long gid = (long long)blockIdx.x * blockDim.x + threadIdx.x;
  long long stride = (long long)gridDim.x * blockDim.x;
  const float L2_099 = -0.0144995696951150f;  // log2(0.99)
  for (; gid < n; gid += stride) {
    float cnt = out[gid];                 // out currently holds pair counts
    float p = exp2f(cnt * L2_099);        // 0.99^cnt
    out[gid] = adj[gid] * p + (1.0f - p) * 100.0f;  // (1-p)/(1-0.99)
  }
}

__global__ void loss_final_kernel(const float* __restrict__ scal, float* __restrict__ out) {
  out[0] = 1.25f * (scal[2] + scal[3]) / ((float)N_ROWS * (float)DDIM);
}

// ------------------------------------------------------------------ launch
extern "C" void kernel_launch(void* const* d_in, const int* in_sizes, int n_in,
                              void* d_out, int out_size, void* d_ws, size_t ws_size,
                              hipStream_t stream) {
  const float* zfr = (const float*)d_in[0];
  const float* zfi = (const float*)d_in[1];
  const float* zsr = (const float*)d_in[2];
  const float* zsi = (const float*)d_in[3];
  const int* prev_syn = (const int*)d_in[4];
  const int* prev_sem = (const int*)d_in[5];
  const float* cb_syn = (const float*)d_in[6];
  const float* cb_sem = (const float*)d_in[7];
  const float* cl_syn = (const float*)d_in[8];
  const float* avg_syn = (const float*)d_in[9];
  const float* cl_sem = (const float*)d_in[10];
  const float* avg_sem = (const float*)d_in[11];
  const float* adj_syn = (const float*)d_in[12];
  const float* adj_sem = (const float*)d_in[13];

  float* out = (float*)d_out;  // flat tuple layout, element offsets:
  float* zq_syn     = out + 0LL;
  float* zq_sem     = out + 16777216LL;
  float* loss       = out + 33554432LL;
  int*   idx_syn    = (int*)(out + 33554433LL);
  int*   idx_sem    = (int*)(out + 33587201LL);
  float* cbout_syn  = out + 33619969LL;
  float* cbout_sem  = out + 34144257LL;
  float* adjout_syn = out + 36241409LL;
  float* adjout_sem = out + 37289985LL;

  unsigned short* cbbf_syn = (unsigned short*)d_ws;
  unsigned short* cbbf_sem = cbbf_syn + (size_t)KSYN * DDIM;
  float* fws = (float*)(cbbf_sem + (size_t)KSEM * DDIM);
  float* cnorm_syn = fws;  fws += KSYN;
  float* cnorm_sem = fws;  fws += KSEM;
  float* counts_syn = fws; fws += KSYN;
  float* counts_sem = fws; fws += KSEM;
  float* clnew_syn = fws;  fws += KSYN;
  float* clnew_sem = fws;  fws += KSEM;
  float* scal = fws;  // [0]=n_syn [1]=n_sem [2]=loss_syn [3]=loss_sem

  // zero accumulation targets (out cb/adj regions double as embed_sum/pair_counts)
  zero_kernel<<<2048, 256, 0, stream>>>(cbout_syn, (long long)KSYN * DDIM);
  zero_kernel<<<2048, 256, 0, stream>>>(cbout_sem, (long long)KSEM * DDIM);
  zero_kernel<<<4096, 256, 0, stream>>>(adjout_syn, (long long)KSYN * KSYN);
  zero_kernel<<<8192, 256, 0, stream>>>(adjout_sem, (long long)KSEM * KSEM);
  zero_kernel<<<64, 256, 0, stream>>>(counts_syn,
      (long long)(KSYN + KSEM + KSYN + KSEM + 4));  // counts..clnew..scal contiguous

  cb_convert_kernel<<<KSYN / 8, 256, 0, stream>>>(cb_syn, cbbf_syn, cnorm_syn, KSYN);
  cb_convert_kernel<<<KSEM / 8, 256, 0, stream>>>(cb_sem, cbbf_sem, cnorm_sem, KSEM);

  vq_argmin_kernel<<<N_ROWS / 128, 256, 0, stream>>>(zfr, zfi, cbbf_syn, cnorm_syn, KSYN, idx_syn);
  vq_argmin_kernel<<<N_ROWS / 128, 256, 0, stream>>>(zsr, zsi, cbbf_sem, cnorm_sem, KSEM, idx_sem);

  row_update_kernel<<<N_ROWS, 128, 0, stream>>>(zfr, zfi, cb_syn, idx_syn, prev_syn, KSYN,
                                                zq_syn, cbout_syn, counts_syn, adjout_syn, scal + 2);
  row_update_kernel<<<N_ROWS, 128, 0, stream>>>(zsr, zsi, cb_sem, idx_sem, prev_sem, KSEM,
                                                zq_sem, cbout_sem, counts_sem, adjout_sem, scal + 3);

  cl_kernel<<<(KSYN + 255) / 256, 256, 0, stream>>>(cl_syn, counts_syn, clnew_syn, scal + 0, KSYN);
  cl_kernel<<<(KSEM + 255) / 256, 256, 0, stream>>>(cl_sem, counts_sem, clnew_sem, scal + 1, KSEM);

  cb_final_kernel<<<(KSYN * DDIM + 255) / 256, 256, 0, stream>>>(avg_syn, clnew_syn, scal + 0, cbout_syn, KSYN);
  cb_final_kernel<<<(KSEM * DDIM + 255) / 256, 256, 0, stream>>>(avg_sem, clnew_sem, scal + 1, cbout_sem, KSEM);

  adj_final_kernel<<<4096, 256, 0, stream>>>(adj_syn, adjout_syn, (long long)KSYN * KSYN);
  adj_final_kernel<<<8192, 256, 0, stream>>>(adj_sem, adjout_sem, (long long)KSEM * KSEM);

  loss_final_kernel<<<1, 1, 0, stream>>>(scal, loss);
}
